// LSTM_11407433138274
// MI455X (gfx1250) — matrix operand order repaired and checked by reading, not compiled, not run on hardware
//
#include <hip/hip_runtime.h>
#include <hip/hip_bf16.h>

// LSTM on MI455X (gfx1250): bf16 WMMA (f32 accumulate) recurrence.
// N=64, T=1024, D=H=1024.
//
// ws layout:
//   [0,        8MB)   WxT : bf16, (4H x D), transposed Wx
//   [8MB,     16MB)   WhT : bf16, (4H x H), transposed Wh
//   [16MB,   +128KB)  h buffer A (bf16, N x H)
//   [...,    +128KB)  h buffer B (bf16, N x H)
//   [...,    +256KB)  c state   (f32,  N x H)
//   [16.5MB, +128MB)  OPTIONAL x in bf16 (N,T,D) — used iff ws_size permits

typedef __bf16 bf16_t;
typedef __attribute__((ext_vector_type(16))) __bf16 v16bf;
typedef __attribute__((ext_vector_type(8)))  __bf16 v8bf;
typedef __attribute__((ext_vector_type(8)))  float  v8f;
typedef __attribute__((ext_vector_type(4)))  float  v4f;

#define NB 64
#define TT 1024
#define DD 1024
#define HH 1024

static __device__ __forceinline__ float sigm_(float v) {
    return 1.0f / (1.0f + __expf(-v));
}
static __device__ __forceinline__ float tanh_(float v) {
    // 1 - 2/(e^{2v}+1): saturates correctly at +-1 for large |v|
    return 1.0f - 2.0f / (__expf(2.0f * v) + 1.0f);
}

// A-operand loaders (ISA 7.12.2 16-bit A layout; p points at row + k + kOff)
static __device__ __forceinline__ v16bf loadA_bf(const bf16_t* p) {
    v8bf a0 = *(const v8bf*)p;
    v8bf a1 = *(const v8bf*)(p + 16);
    return __builtin_shufflevector(a0, a1,
        0, 1, 2, 3, 4, 5, 6, 7, 8, 9, 10, 11, 12, 13, 14, 15);
}
static __device__ __forceinline__ v16bf loadA_f32(const float* p) {
    v4f x0 = *(const v4f*)p;
    v4f x1 = *(const v4f*)(p + 4);
    v4f x2 = *(const v4f*)(p + 16);
    v4f x3 = *(const v4f*)(p + 20);
    v16bf a;
#pragma unroll
    for (int j = 0; j < 4; ++j) {
        a[j]      = (bf16_t)x0[j];
        a[4 + j]  = (bf16_t)x1[j];
        a[8 + j]  = (bf16_t)x2[j];
        a[12 + j] = (bf16_t)x3[j];
    }
    return a;
}

// ---------------------------------------------------------------------------
// Prep: transpose + convert f32 (1024 x 4096) -> bf16 (4096 x 1024).
// blockIdx.z selects Wx (0) or Wh (1). Tiled through LDS for coalescing.
// ---------------------------------------------------------------------------
__global__ __launch_bounds__(256) void lstm_prep_transpose(
    const float* __restrict__ Wx, const float* __restrict__ Wh,
    bf16_t* __restrict__ WxT, bf16_t* __restrict__ WhT)
{
    __shared__ float tile[32][33];
    const float*  W  = blockIdx.z ? Wh  : Wx;
    bf16_t*       WT = blockIdx.z ? WhT : WxT;
    const int c0 = blockIdx.x * 32;   // column in 4H
    const int k0 = blockIdx.y * 32;   // row    in D/H
    const int tx = threadIdx.x;       // 0..31
    const int ty = threadIdx.y;       // 0..7
#pragma unroll
    for (int j = 0; j < 32; j += 8)
        tile[ty + j][tx] = W[(size_t)(k0 + ty + j) * (4 * HH) + (c0 + tx)];
    __syncthreads();
#pragma unroll
    for (int j = 0; j < 32; j += 8)
        WT[(size_t)(c0 + ty + j) * DD + (k0 + tx)] = (bf16_t)tile[tx][ty + j];
}

// ---------------------------------------------------------------------------
// Prep: x (f32) -> x (bf16), straight streaming copy. 8 elems/thread.
// ---------------------------------------------------------------------------
__global__ __launch_bounds__(256) void lstm_prep_xbf16(
    const float* __restrict__ x, bf16_t* __restrict__ xbf)
{
    const size_t i8 = ((size_t)blockIdx.x * 256 + threadIdx.x) * 8;
    v4f a = *(const v4f*)(x + i8);
    v4f b = *(const v4f*)(x + i8 + 4);
    v8bf o;
#pragma unroll
    for (int j = 0; j < 4; ++j) {
        o[j]     = (bf16_t)a[j];
        o[4 + j] = (bf16_t)b[j];
    }
    *(v8bf*)(xbf + i8) = o;
}

// ---------------------------------------------------------------------------
// Init: c = 0, h_bf16 = bf16(h0)
// ---------------------------------------------------------------------------
__global__ __launch_bounds__(256) void lstm_init(
    const float* __restrict__ h0, bf16_t* __restrict__ hbf, float* __restrict__ c)
{
    int i = blockIdx.x * blockDim.x + threadIdx.x;
    if (i < NB * HH) {
        c[i]   = 0.0f;
        hbf[i] = (bf16_t)h0[i];
    }
}

// ---------------------------------------------------------------------------
// One timestep: A = x_t@Wx + h@Wh + b (bf16 WMMA, f32 accum), gates, c/h.
// grid = H/16 blocks (one per 16-col block of H); block = 512 thr = 16 waves:
//   wave w: M-tile mt = w&3 (rows 16*mt..+15), K-split ks = w>>2.
// Each wave computes ALL 4 gate tiles over a quarter of the K range,
// reusing the A chunk from registers (10 b128 loads per 4 WMMAs); the 4
// K-split partial sums are reduced in LDS. 4 waves/SIMD hide load latency.
// XBF: x already converted to bf16 (preferred; pure b128-load A path).
// ---------------------------------------------------------------------------
template <bool XBF>
__global__ __launch_bounds__(512) void lstm_step(
    const float*  __restrict__ x,    const bf16_t* __restrict__ xbf,
    const float*  __restrict__ bias,
    const bf16_t* __restrict__ WxT,  const bf16_t* __restrict__ WhT,
    const bf16_t* __restrict__ hIn,  bf16_t* __restrict__ hOut,
    float* __restrict__ cSt, float* __restrict__ out, int t)
{
    __shared__ float sA[4][4][64][17]; // [gate][ksplit][M][n] (+1 pad)

    const int tid  = threadIdx.x;
    const int w    = tid >> 5;        // wave 0..15 (wave32)
    const int mt   = w & 3;           // M tile
    const int ks   = w >> 2;          // K split 0..3
    const int l    = tid & 31;        // lane
    const int n    = l & 15;          // tile column (N) / A row within tile
    const int hi   = l >> 4;          // half-wave select
    const int kOff = hi * 8;          // A-layout K offset (ISA 7.12.2)
    const int bOff = hi * 8;          // B-layout K offset
    const int m    = mt * 16 + n;                   // batch row for A
    const int colH = blockIdx.x * 16 + n;           // column within H
    const int kBeg = ks * (DD / 4);                 // this wave's K range
    const int kEnd = kBeg + (DD / 4);

    // --- 4 partial accumulators (one per gate); bias added at reduction ----
    v8f acc[4];
#pragma unroll
    for (int gi = 0; gi < 4; ++gi)
#pragma unroll
        for (int j = 0; j < 8; ++j) acc[gi][j] = 0.0f;

    // ---------------- x_t @ Wx  (K in [kBeg,kEnd)) -------------------------
    {
        const bf16_t* wxg0 = WxT + ((size_t)(0 * HH) + colH) * DD + bOff;
        const bf16_t* wxg1 = WxT + ((size_t)(1 * HH) + colH) * DD + bOff;
        const bf16_t* wxg2 = WxT + ((size_t)(2 * HH) + colH) * DD + bOff;
        const bf16_t* wxg3 = WxT + ((size_t)(3 * HH) + colH) * DD + bOff;
        const float*  xrF  = x   + (size_t)m * TT * DD + (size_t)t * DD + kOff;
        const bf16_t* xrB  = xbf + (size_t)m * TT * DD + (size_t)t * DD + kOff;
#pragma unroll 2
        for (int k0 = kBeg; k0 < kEnd; k0 += 32) {
            v16bf a  = XBF ? loadA_bf(xrB + k0) : loadA_f32(xrF + k0);
            v16bf b0 = loadA_bf(wxg0 + k0);
            v16bf b1 = loadA_bf(wxg1 + k0);
            v16bf b2 = loadA_bf(wxg2 + k0);
            v16bf b3 = loadA_bf(wxg3 + k0);
            acc[0] = __builtin_amdgcn_wmma_f32_16x16x32_bf16(
                false, a, false, b0, (short)0, acc[0], false, false);
            acc[1] = __builtin_amdgcn_wmma_f32_16x16x32_bf16(
                false, a, false, b1, (short)0, acc[1], false, false);
            acc[2] = __builtin_amdgcn_wmma_f32_16x16x32_bf16(
                false, a, false, b2, (short)0, acc[2], false, false);
            acc[3] = __builtin_amdgcn_wmma_f32_16x16x32_bf16(
                false, a, false, b3, (short)0, acc[3], false, false);
        }
    }

    // ---------------- h @ Wh  (K in [kBeg,kEnd), bf16 h state) -------------
    {
        const bf16_t* whg0 = WhT + ((size_t)(0 * HH) + colH) * HH + bOff;
        const bf16_t* whg1 = WhT + ((size_t)(1 * HH) + colH) * HH + bOff;
        const bf16_t* whg2 = WhT + ((size_t)(2 * HH) + colH) * HH + bOff;
        const bf16_t* whg3 = WhT + ((size_t)(3 * HH) + colH) * HH + bOff;
        const bf16_t* hrow = hIn + (size_t)m * HH + kOff;
#pragma unroll 2
        for (int k0 = kBeg; k0 < kEnd; k0 += 32) {
            v16bf a  = loadA_bf(hrow + k0);
            v16bf b0 = loadA_bf(whg0 + k0);
            v16bf b1 = loadA_bf(whg1 + k0);
            v16bf b2 = loadA_bf(whg2 + k0);
            v16bf b3 = loadA_bf(whg3 + k0);
            acc[0] = __builtin_amdgcn_wmma_f32_16x16x32_bf16(
                false, a, false, b0, (short)0, acc[0], false, false);
            acc[1] = __builtin_amdgcn_wmma_f32_16x16x32_bf16(
                false, a, false, b1, (short)0, acc[1], false, false);
            acc[2] = __builtin_amdgcn_wmma_f32_16x16x32_bf16(
                false, a, false, b2, (short)0, acc[2], false, false);
            acc[3] = __builtin_amdgcn_wmma_f32_16x16x32_bf16(
                false, a, false, b3, (short)0, acc[3], false, false);
        }
    }

    // --- prefetch next timestep's x row into L2 (serial-scan latency hide) -
    // Block bx prefetches batch row bx (gridDim.x == NB == 64). 128B lines.
    if (t + 1 < TT) {
        const int rowBytes = XBF ? DD * 2 : DD * 4;
        if (tid * 128 < rowBytes) {
            const char* p = XBF
                ? (const char*)(xbf + (size_t)blockIdx.x * TT * DD + (size_t)(t + 1) * DD)
                : (const char*)(x   + (size_t)blockIdx.x * TT * DD + (size_t)(t + 1) * DD);
            __builtin_prefetch(p + tid * 128, 0, 1);
        }
    }

    // --- stage partial sums in LDS (C/D layout: M = r + 8*hi) --------------
#pragma unroll
    for (int gi = 0; gi < 4; ++gi)
#pragma unroll
        for (int r = 0; r < 8; ++r)
            sA[gi][ks][mt * 16 + r + hi * 8][n] = acc[gi][r];
    __syncthreads();

    // --- K-split reduce + bias + gates: 64x16 elements, 2 per thread -------
    for (int e = tid; e < NB * 16; e += 512) {
        const int em = e >> 4;
        const int en = e & 15;
        const int gc = blockIdx.x * 16 + en;        // column in H
        float g4[4];
#pragma unroll
        for (int gi = 0; gi < 4; ++gi)
            g4[gi] = (sA[gi][0][em][en] + sA[gi][1][em][en])
                   + (sA[gi][2][em][en] + sA[gi][3][em][en])
                   + bias[gi * HH + gc];
        const size_t ci = (size_t)em * HH + gc;
        const float cOld = cSt[ci];
        const float cNew = sigm_(g4[1]) * cOld + sigm_(g4[0]) * tanh_(g4[3]);
        const float hNew = sigm_(g4[2]) * tanh_(cNew);
        cSt[ci]  = cNew;
        hOut[ci] = (bf16_t)hNew;
        out[(size_t)em * TT * HH + (size_t)t * HH + gc] = hNew;
    }
}

// ---------------------------------------------------------------------------
extern "C" void kernel_launch(void* const* d_in, const int* in_sizes, int n_in,
                              void* d_out, int out_size, void* d_ws, size_t ws_size,
                              hipStream_t stream)
{
    const float* x  = (const float*)d_in[0];   // (N,T,D)
    const float* h0 = (const float*)d_in[1];   // (N,H)
    const float* Wx = (const float*)d_in[2];   // (D,4H)
    const float* Wh = (const float*)d_in[3];   // (H,4H)
    const float* b  = (const float*)d_in[4];   // (4H,)
    float* out = (float*)d_out;                // (N,T,H)

    char* ws = (char*)d_ws;
    bf16_t* WxT = (bf16_t*)ws;                                   // 8 MB
    bf16_t* WhT = (bf16_t*)(ws + ((size_t)8 << 20));             // 8 MB
    bf16_t* hA  = (bf16_t*)(ws + ((size_t)16 << 20));            // 128 KB
    bf16_t* hB  = hA + (size_t)NB * HH;                          // 128 KB
    float*  c   = (float*)(ws + ((size_t)16 << 20)
                              + (size_t)2 * NB * HH * sizeof(bf16_t)); // 256 KB
    const size_t baseBytes = ((size_t)16 << 20)
                           + (size_t)2 * NB * HH * sizeof(bf16_t)
                           + (size_t)NB * HH * sizeof(float);
    const size_t xbfBytes  = (size_t)NB * TT * DD * sizeof(bf16_t);   // 128 MB
    const bool useXbf = (ws_size >= baseBytes + xbfBytes);
    bf16_t* xbf = (bf16_t*)(ws + baseBytes);

    // 1) transpose+convert both weight matrices to bf16 (4H x K)
    {
        dim3 blk(32, 8);
        dim3 grd((4 * HH) / 32, DD / 32, 2);
        lstm_prep_transpose<<<grd, blk, 0, stream>>>(Wx, Wh, WxT, WhT);
    }
    // 2) init c=0, h=bf16(h0)
    lstm_init<<<(NB * HH + 255) / 256, 256, 0, stream>>>(h0, hA, c);
    // 3) optional: one-time x -> bf16 (hoists cvt out of the serial chain)
    if (useXbf) {
        const size_t nElem = (size_t)NB * TT * DD;         // 2048 per block
        lstm_prep_xbf16<<<(unsigned)(nElem / 2048), 256, 0, stream>>>(x, xbf);
    }

    // 4) sequential scan: one kernel per timestep (kernel boundary = global
    //    sync; h state double-buffered to avoid cross-block read/write race)
    for (int t = 0; t < TT; ++t) {
        bf16_t* hin  = (t & 1) ? hB : hA;
        bf16_t* hout = (t & 1) ? hA : hB;
        if (useXbf)
            lstm_step<true><<<HH / 16, 512, 0, stream>>>(
                x, xbf, b, WxT, WhT, hin, hout, c, out, t);
        else
            lstm_step<false><<<HH / 16, 512, 0, stream>>>(
                x, xbf, b, WxT, WhT, hin, hout, c, out, t);
    }
}
